// Str2Str_42356967473475
// MI455X (gfx1250) — matrix operand order repaired and checked by reading, not compiled, not run on hardware
//
#include <hip/hip_runtime.h>
#include <hip/hip_bf16.h>
#include <math.h>

// ---------------------------------------------------------------------------
// MI455X (gfx1250, wave32) Str2Str structure update.
// - top-K first => edge/message net only on 768x128 selected pairs
// - all GEMMs via v_wmma_f32_16x16x32_bf16, A staged bf16 in LDS
// - pair-row gathers + s0 staging via global_load_async_to_lds_b128
// ---------------------------------------------------------------------------

typedef __bf16 bf16;
typedef __attribute__((ext_vector_type(16))) __bf16 v16bf;
typedef __attribute__((ext_vector_type(8)))  float  v8f;

#define LSEQ 768
#define TOPK 128

// --------------------------- async-to-LDS helpers --------------------------

__device__ __forceinline__ void async_copy_b128(void* lds_dst, const void* gsrc) {
  unsigned l = (unsigned)(uintptr_t)lds_dst;      // low 32b of shared ptr = LDS addr
  unsigned long long g = (unsigned long long)(uintptr_t)gsrc;
  asm volatile("global_load_async_to_lds_b128 %0, %1, off"
               :: "v"(l), "v"(g) : "memory");
}
__device__ __forceinline__ void wait_async0() {
  asm volatile("s_wait_asynccnt 0x0" ::: "memory");
}

// ------------------------------ WMMA helpers -------------------------------

__device__ __forceinline__ v8f wmma_bf16(v16bf a, v16bf b, v8f c) {
  return __builtin_amdgcn_wmma_f32_16x16x32_bf16(
      false, a, false, b, (short)0, c, false, false);
}

// A-fragment (16x32 bf16) from bf16 LDS tile, ISA layout:
//   lane L: m = L%16 ; element e -> k = kk + e + (e>=8 ? 8:0) + (L>=16 ? 8:0)
// Two contiguous 8-element (16B) runs per lane.
__device__ __forceinline__ v16bf load_a_frag(const bf16* lds, int stride,
                                             int kk, int lane) {
  int m  = lane & 15;
  int hi = (lane >> 4) << 3;
  const bf16* p = lds + m * stride + kk + hi;
  v16bf a;
#pragma unroll
  for (int e = 0; e < 8; ++e) a[e] = p[e];
#pragma unroll
  for (int e = 8; e < 16; ++e) a[e] = p[e + 8];
  return a;
}

__device__ __forceinline__ v16bf load_b_frag(const bf16* fb, int lane) {
  v16bf b;
  const bf16* p = fb + lane * 16;
#pragma unroll
  for (int e = 0; e < 16; ++e) b[e] = p[e];
  return b;
}

// C(16 x ntiles*16) = A(16 x kchunks*32) @ B + bias (+ init); one wave.
// Writes f32 C (if ldsC) and/or bf16 C (if ldsCbf).
__device__ __forceinline__ void wave_gemm(
    const bf16* ldsA, int strideA, int kchunks,
    const unsigned short* fragB, int ntiles,
    const float* bias, int biasN,
    const float* ldsInit, int strideInit,
    float* ldsC, bf16* ldsCbf, int strideC, int lane, bool relu_out) {
  const bf16* B = (const bf16*)fragB;
  int nloc = lane & 15;
  int mhi  = (lane >> 4) << 3;
  for (int t = 0; t < ntiles; ++t) {
    int n = t * 16 + nloc;
    float bv = (bias != nullptr && n < biasN) ? bias[n] : 0.0f;
    v8f c;
#pragma unroll
    for (int r = 0; r < 8; ++r)
      c[r] = bv + (ldsInit ? ldsInit[(r + mhi) * strideInit + n] : 0.0f);
    for (int kc = 0; kc < kchunks; ++kc) {
      v16bf a = load_a_frag(ldsA, strideA, kc * 32, lane);
      v16bf b = load_b_frag(B + (kc * ntiles + t) * 512, lane);
      c = wmma_bf16(a, b, c);
    }
#pragma unroll
    for (int r = 0; r < 8; ++r) {
      float v = c[r];
      if (relu_out) v = fmaxf(v, 0.0f);
      if (ldsC)   ldsC[(r + mhi) * strideC + n] = v;
      if (ldsCbf) ldsCbf[(r + mhi) * strideC + n] = (bf16)v;
    }
  }
}

// ------------------------------ LN helpers ---------------------------------

// LayerNorm of one Len-vector (global or LDS f32) -> bf16 dst, whole wave.
__device__ __forceinline__ void wave_vec_ln_bf(const float* g, int Len,
                                               bf16* dst, int lane) {
  float s = 0.0f;
  for (int c = lane; c < Len; c += 32) s += g[c];
#pragma unroll
  for (int m = 16; m; m >>= 1) s += __shfl_xor(s, m);
  float mean = s / (float)Len;
  float v = 0.0f;
  for (int c = lane; c < Len; c += 32) { float d = g[c] - mean; v += d * d; }
#pragma unroll
  for (int m = 16; m; m >>= 1) v += __shfl_xor(v, m);
  float inv = rsqrtf(v / (float)Len + 1e-5f);
  for (int c = lane; c < Len; c += 32) dst[c] = (bf16)((g[c] - mean) * inv);
}

// Row LN of 16 x N f32 tile -> f32 dst
__device__ __forceinline__ void wave_row_ln(const float* src, int ss,
                                            float* dst, int ds, int N,
                                            int lane) {
  int row = lane >> 1;
  int c0  = (lane & 1) * (N >> 1);
  float s = 0.0f;
  for (int c = 0; c < (N >> 1); ++c) s += src[row * ss + c0 + c];
  s += __shfl_xor(s, 1);
  float mean = s / (float)N;
  float v = 0.0f;
  for (int c = 0; c < (N >> 1); ++c) {
    float d = src[row * ss + c0 + c] - mean; v += d * d;
  }
  v += __shfl_xor(v, 1);
  float inv = rsqrtf(v / (float)N + 1e-5f);
  for (int c = 0; c < (N >> 1); ++c)
    dst[row * ds + c0 + c] = (src[row * ss + c0 + c] - mean) * inv;
}

// Row LN of 16 x N f32 tile -> bf16 dst
__device__ __forceinline__ void wave_row_ln_bf(const float* src, int ss,
                                               bf16* dst, int ds, int N,
                                               int lane) {
  int row = lane >> 1;
  int c0  = (lane & 1) * (N >> 1);
  float s = 0.0f;
  for (int c = 0; c < (N >> 1); ++c) s += src[row * ss + c0 + c];
  s += __shfl_xor(s, 1);
  float mean = s / (float)N;
  float v = 0.0f;
  for (int c = 0; c < (N >> 1); ++c) {
    float d = src[row * ss + c0 + c] - mean; v += d * d;
  }
  v += __shfl_xor(v, 1);
  float inv = rsqrtf(v / (float)N + 1e-5f);
  for (int c = 0; c < (N >> 1); ++c)
    dst[row * ds + c0 + c] = (bf16)((src[row * ss + c0 + c] - mean) * inv);
}

// --------------------------- weight pre-packing ----------------------------

__global__ void pack_b_kernel(const float* __restrict__ W,
                              unsigned short* __restrict__ dst_u16,
                              int K, int N, int Kpad, int Npad) {
  bf16* dst = (bf16*)dst_u16;
  int tid = blockIdx.x * blockDim.x + threadIdx.x;
  int total = Kpad * Npad;
  if (tid >= total) return;
  int fb = tid >> 9;
  int r  = tid & 511;
  int lane = r >> 4, e = r & 15;
  int ntiles = Npad >> 4;
  int chunk = fb / ntiles, t = fb % ntiles;
  int n = t * 16 + (lane & 15);
  int k = chunk * 32 + (e & 7) + ((lane >> 4) << 3) + ((e >> 3) << 4);
  float v = (k < K && n < N) ? W[k * N + n] : 0.0f;
  dst[tid] = (bf16)v;
}

// ------------------------------ node kernel --------------------------------

__global__ __launch_bounds__(32) void node_kernel(
    const float* __restrict__ msa, const float* __restrict__ state,
    const unsigned short* fWen, const unsigned short* fWn1,
    const unsigned short* fWn2,
    const float* b_en, const float* bn1, const float* bn2,
    unsigned short* __restrict__ s0_bf_u, float* __restrict__ node_out,
    unsigned short* __restrict__ node_bf_u) {
  __shared__ bf16  A[16 * 288];
  __shared__ float nbuf[16 * 32];
  __shared__ float tbuf[16 * 32];
  __shared__ bf16  tbf[16 * 32];
  __shared__ bf16  hbf[16 * 64];
  bf16* s0_bf = (bf16*)s0_bf_u;
  bf16* node_bf = (bf16*)node_bf_u;
  int lane = threadIdx.x;
  int i0 = blockIdx.x * 16;
  for (int r = 0; r < 16; ++r) {
    int i = i0 + r;
    wave_vec_ln_bf(msa + (size_t)i * 256, 256, A + r * 288, lane);
    for (int c = lane; c < 256; c += 32) s0_bf[(size_t)i * 256 + c] = A[r * 288 + c];
    wave_vec_ln_bf(state + i * 16, 16, A + r * 288 + 256, lane);
    if (lane < 16) A[r * 288 + 272 + lane] = (bf16)0.0f;
  }
  __syncthreads();
  wave_gemm(A, 288, 9, fWen, 2, b_en, 32, nullptr, 0, nbuf, nullptr, 32, lane, false);
  __syncthreads();
  wave_row_ln_bf(nbuf, 32, tbf, 32, 32, lane);
  __syncthreads();
  wave_gemm(tbf, 32, 1, fWn1, 4, bn1, 64, nullptr, 0, nullptr, hbf, 64, lane, true);
  __syncthreads();
  wave_gemm(hbf, 64, 2, fWn2, 2, bn2, 32, nbuf, 32, nbuf, nullptr, 32, lane, false);
  __syncthreads();
  wave_row_ln(nbuf, 32, tbuf, 32, 32, lane);
  __syncthreads();
  for (int r = 0; r < 16; ++r) {
    float v = tbuf[r * 32 + lane];
    node_out[(i0 + r) * 32 + lane] = v;
    node_bf[(i0 + r) * 32 + lane] = (bf16)v;
  }
}

// ------------------------------ top-K kernel -------------------------------

__global__ __launch_bounds__(256) void topk_kernel(
    const float* __restrict__ xyz, int* __restrict__ nbr,
    float* __restrict__ Dsel) {
  __shared__ float drow[LSEQ];
  __shared__ float rv[256];
  __shared__ int   ri[256];
  int i = blockIdx.x, t = threadIdx.x;
  float cx = xyz[i * 9 + 3], cy = xyz[i * 9 + 4], cz = xyz[i * 9 + 5];
  for (int j = t; j < LSEQ; j += 256) {
    float dx = xyz[j * 9 + 3] - cx, dy = xyz[j * 9 + 4] - cy,
          dz = xyz[j * 9 + 5] - cz;
    drow[j] = sqrtf(dx * dx + dy * dy + dz * dz + 1e-8f);
  }
  __syncthreads();
  for (int k = 0; k < TOPK; ++k) {
    float bv = 3.4e38f; int bi = 0x7fffffff;
    for (int j = t; j < LSEQ; j += 256) {
      float v = drow[j];
      if (v < bv || (v == bv && j < bi)) { bv = v; bi = j; }
    }
    rv[t] = bv; ri[t] = bi;
    __syncthreads();
    for (int s = 128; s; s >>= 1) {
      if (t < s) {
        float ov = rv[t + s]; int oi = ri[t + s];
        if (ov < rv[t] || (ov == rv[t] && oi < ri[t])) { rv[t] = ov; ri[t] = oi; }
      }
      __syncthreads();
    }
    if (t == 0) {
      nbr[i * TOPK + k] = ri[0];
      Dsel[i * TOPK + k] = rv[0];
      drow[ri[0]] = 3.4e38f;
    }
    __syncthreads();
  }
}

// --------------------------- edge + message kernel -------------------------
// One wave per (residue i, tile of 16 neighbors). Pair rows arrive via
// global_load_async_to_lds_b128 (16B/lane, one op per row, all 16 in flight).

__global__ __launch_bounds__(32) void edge_msg_kernel(
    const float* __restrict__ pair, const int* __restrict__ idx,
    const float* __restrict__ xyz, const unsigned short* __restrict__ node_bf_u,
    const int* __restrict__ nbr, const float* __restrict__ Dsel,
    const unsigned short* fWee, const unsigned short* fWe1,
    const unsigned short* fWe2, const unsigned short* fWm1,
    const unsigned short* fWm2,
    const float* b_ee, const float* be1, const float* be2,
    const float* bm1, const float* bm2,
    float* __restrict__ partial) {
  __shared__ float praw[16 * 128];   // async-staged pair rows
  __shared__ bf16  A[16 * 224];
  __shared__ float ebuf[16 * 32];
  __shared__ bf16  tbf[16 * 32];
  __shared__ bf16  hbf[16 * 64];
  __shared__ bf16  minb[16 * 96];
  __shared__ bf16  mbf[16 * 64];
  __shared__ float obuf[16 * 48];
  __shared__ float relb[16 * 3];
  __shared__ int   jrow[16];
  const bf16* node_bf = (const bf16*)node_bf_u;
  int lane = threadIdx.x;
  int i = blockIdx.x, tt = blockIdx.y;
  float cx = xyz[i * 9 + 3], cy = xyz[i * 9 + 4], cz = xyz[i * 9 + 5];
  int ii = idx[i];

  // issue all 16 gathered row copies asynchronously into LDS
  for (int r = 0; r < 16; ++r) {
    int j = nbr[i * TOPK + tt * 16 + r];
    if (lane == 0) jrow[r] = j;
    async_copy_b128(&praw[r * 128 + lane * 4],
                    pair + ((size_t)i * LSEQ + j) * 128 + lane * 4);
  }
  wait_async0();
  __syncthreads();

  for (int r = 0; r < 16; ++r) {
    wave_vec_ln_bf(praw + r * 128, 128, A + r * 224, lane);
    float d = Dsel[i * TOPK + tt * 16 + r];
    for (int c = lane; c < 64; c += 32) {
      float mu = 2.0f + 20.0f * (float)c / 63.0f;
      float u = (d - mu) * 3.2f;  // 1/sigma, sigma = 20/64
      A[r * 224 + 128 + c] = (bf16)expf(-u * u);
    }
    int j = jrow[r];
    if (lane == 0) {
      float off = (float)(idx[j] - ii);
      float sg = (off > 0.0f) ? 1.0f : ((off < 0.0f) ? -1.0f : 0.0f);
      A[r * 224 + 192] = (bf16)(sg * logf(fabsf(off) + 1.0f));
      relb[r * 3 + 0] = xyz[j * 9 + 3] - cx;
      relb[r * 3 + 1] = xyz[j * 9 + 4] - cy;
      relb[r * 3 + 2] = xyz[j * 9 + 5] - cz;
    }
    if (lane < 31) A[r * 224 + 193 + lane] = (bf16)0.0f;  // pad 193..223
  }
  __syncthreads();
  wave_gemm(A, 224, 7, fWee, 2, b_ee, 32, nullptr, 0, ebuf, nullptr, 32, lane, false);
  __syncthreads();
  wave_row_ln_bf(ebuf, 32, tbf, 32, 32, lane);
  __syncthreads();
  wave_gemm(tbf, 32, 1, fWe1, 4, be1, 64, nullptr, 0, nullptr, hbf, 64, lane, true);
  __syncthreads();
  wave_gemm(hbf, 64, 2, fWe2, 2, be2, 32, ebuf, 32, ebuf, nullptr, 32, lane, false);
  __syncthreads();
  wave_row_ln_bf(ebuf, 32, minb + 64, 96, 32, lane);  // edge -> m_in[64:96]
  for (int r = 0; r < 16; ++r) {
    minb[r * 96 + lane]      = node_bf[i * 32 + lane];        // h_i
    minb[r * 96 + 32 + lane] = node_bf[jrow[r] * 32 + lane];  // h_j
  }
  __syncthreads();
  wave_gemm(minb, 96, 3, fWm1, 4, bm1, 64, nullptr, 0, nullptr, mbf, 64, lane, true);
  __syncthreads();
  wave_gemm(mbf, 64, 2, fWm2, 3, bm2, 40, nullptr, 0, obuf, nullptr, 48, lane, false);
  __syncthreads();
  // deterministic per-tile partials: [0:32]=m_l0 sum, [32:38]=w_atom sum,
  // [38:44]=vec_edge sum
  float* P = partial + ((size_t)i * 8 + tt) * 48;
  {
    float s = 0.0f;
    for (int r = 0; r < 16; ++r) s += obuf[r * 48 + lane];
    P[lane] = s;
  }
  if (lane < 6) {
    float w = 0.0f;
    for (int r = 0; r < 16; ++r) w += obuf[r * 48 + 34 + lane];
    P[32 + lane] = w;
  }
  if (lane >= 8 && lane < 14) {
    int c = (lane - 8) / 3, d = (lane - 8) % 3;
    float v = 0.0f;
    for (int r = 0; r < 16; ++r) v += obuf[r * 48 + 32 + c] * relb[r * 3 + d];
    P[38 + (lane - 8)] = v;
  }
}

// ----------------------------- finalize kernel -----------------------------

__global__ void finalize_kernel(
    const float* __restrict__ partial, const float* __restrict__ node,
    const float* __restrict__ xyz, const unsigned char* __restrict__ rmask,
    const float* __restrict__ Wl0, const float* __restrict__ bl0,
    float* __restrict__ state_out, float* __restrict__ out) {
  int i = blockIdx.x * blockDim.x + threadIdx.x;
  if (i >= LSEQ) return;
  float msum[32], wsum[6], vsum[6];
  for (int c = 0; c < 32; ++c) msum[c] = 0.0f;
  for (int c = 0; c < 6; ++c) { wsum[c] = 0.0f; vsum[c] = 0.0f; }
  for (int t = 0; t < 8; ++t) {
    const float* P = partial + ((size_t)i * 8 + t) * 48;
    for (int c = 0; c < 32; ++c) msum[c] += P[c];
    for (int c = 0; c < 6; ++c) wsum[c] += P[32 + c];
    for (int c = 0; c < 6; ++c) vsum[c] += P[38 + c];
  }
  const float inv = 1.0f / 128.0f;
  for (int o = 0; o < 16; ++o) {
    float acc = bl0[o];
    for (int c = 0; c < 32; ++c) acc += node[i * 32 + c] * Wl0[c * 16 + o];
    for (int c = 0; c < 32; ++c) acc += (msum[c] * inv) * Wl0[(32 + c) * 16 + o];
    state_out[i * 16 + o] = acc;
    out[6912 + i * 16 + o] = acc;
  }
  float ca[3];
  for (int d = 0; d < 3; ++d) ca[d] = xyz[i * 9 + 3 + d];
  float l1[3][3];
  for (int a = 0; a < 3; ++a)
    for (int d = 0; d < 3; ++d) l1[a][d] = xyz[i * 9 + a * 3 + d] - ca[d];
  float off[2][3];
  for (int c = 0; c < 2; ++c)
    for (int d = 0; d < 3; ++d) {
      float va = 0.0f;
      for (int a = 0; a < 3; ++a) va += (wsum[c * 3 + a] * inv) * l1[a][d];
      off[c][d] = vsum[c * 3 + d] * inv + va;
    }
  float T[3], R3[3];
  for (int d = 0; d < 3; ++d) { T[d] = off[0][d] * 0.1f; R3[d] = off[1][d] * 0.01f; }
  float qn = sqrtf(1.0f + R3[0] * R3[0] + R3[1] * R3[1] + R3[2] * R3[2]);
  float qA = 1.0f / qn, qB = R3[0] / qn, qC = R3[1] / qn, qD = R3[2] / qn;
  float Rm[3][3];
  Rm[0][0] = qA * qA + qB * qB - qC * qC - qD * qD;
  Rm[0][1] = 2 * qB * qC - 2 * qA * qD;
  Rm[0][2] = 2 * qB * qD + 2 * qA * qC;
  Rm[1][0] = 2 * qB * qC + 2 * qA * qD;
  Rm[1][1] = qA * qA - qB * qB + qC * qC - qD * qD;
  Rm[1][2] = 2 * qC * qD - 2 * qA * qB;
  Rm[2][0] = 2 * qB * qD - 2 * qA * qC;
  Rm[2][1] = 2 * qC * qD + 2 * qA * qB;
  Rm[2][2] = qA * qA - qB * qB - qC * qC + qD * qD;
  if (rmask[i]) {
    for (int r = 0; r < 3; ++r)
      for (int c = 0; c < 3; ++c) Rm[r][c] = (r == c) ? 1.0f : 0.0f;
  }
  for (int a = 0; a < 3; ++a)
    for (int r = 0; r < 3; ++r) {
      float v = 0.0f;
      for (int j = 0; j < 3; ++j) v += Rm[r][j] * l1[a][j];
      out[i * 9 + a * 3 + r] = v + ca[r] + T[r];
    }
}

// ------------------------------- si kernel ---------------------------------

__global__ __launch_bounds__(32) void si_kernel(
    const unsigned short* __restrict__ s0_bf_u,
    const float* __restrict__ state_out,
    const unsigned short* fWs0, const unsigned short* fWsi,
    const unsigned short* fWp1, const unsigned short* fWp2,
    const unsigned short* fWp3, const unsigned short* fWp4,
    const unsigned short* fWout,
    const float* bs0, const float* bsi,
    const float* bp1, const float* bp2, const float* bp3, const float* bp4,
    const float* bout, float* __restrict__ out) {
  __shared__ bf16  A0[16 * 256];
  __shared__ bf16  Asi[16 * 32];
  __shared__ float si[16 * 128];
  __shared__ bf16  act[16 * 128];
  __shared__ bf16  hbf[16 * 128];
  __shared__ float ab[16 * 32];
  const bf16* s0_bf = (const bf16*)s0_bf_u;
  int lane = threadIdx.x;
  int i0 = blockIdx.x * 16;
  // async-stage the 16x256 bf16 s0 tile (8 bf16 = 16B per lane per row)
  for (int r = 0; r < 16; ++r)
    async_copy_b128(&A0[r * 256 + lane * 8],
                    s0_bf + (size_t)(i0 + r) * 256 + lane * 8);
  for (int r = 0; r < 16; ++r) {
    wave_vec_ln_bf(state_out + (i0 + r) * 16, 16, Asi + r * 32, lane);
    if (lane < 16) Asi[r * 32 + 16 + lane] = (bf16)0.0f;
  }
  wait_async0();
  __syncthreads();
  wave_gemm(A0, 256, 8, fWs0, 8, bs0, 128, nullptr, 0, si, nullptr, 128, lane, false);
  __syncthreads();
  wave_gemm(Asi, 32, 1, fWsi, 8, bsi, 128, si, 128, si, nullptr, 128, lane, false);
  __syncthreads();
  for (int blk = 0; blk < 2; ++blk) {
    const unsigned short* fa = blk ? fWp3 : fWp1;
    const unsigned short* fb = blk ? fWp4 : fWp2;
    const float* ba = blk ? bp3 : bp1;
    const float* bb = blk ? bp4 : bp2;
    for (int z = lane; z < 16 * 128; z += 32) act[z] = (bf16)fmaxf(si[z], 0.0f);
    __syncthreads();
    wave_gemm(act, 128, 4, fa, 8, ba, 128, nullptr, 0, nullptr, hbf, 128, lane, true);
    __syncthreads();
    wave_gemm(hbf, 128, 4, fb, 8, bb, 128, si, 128, si, nullptr, 128, lane, false);
    __syncthreads();
  }
  for (int z = lane; z < 16 * 128; z += 32) act[z] = (bf16)fmaxf(si[z], 0.0f);
  __syncthreads();
  wave_gemm(act, 128, 4, fWout, 2, bout, 20, nullptr, 0, ab, nullptr, 32, lane, false);
  __syncthreads();
  for (int r = 0; r < 16; ++r)
    if (lane < 20) out[19200 + (i0 + r) * 20 + lane] = ab[r * 32 + lane];
}

// ------------------------------ host launcher ------------------------------

extern "C" void kernel_launch(void* const* d_in, const int* in_sizes, int n_in,
                              void* d_out, int out_size, void* d_ws,
                              size_t ws_size, hipStream_t stream) {
  const float* msa   = (const float*)d_in[0];
  const float* pair  = (const float*)d_in[1];
  const float* xyz   = (const float*)d_in[2];
  const float* state = (const float*)d_in[3];
  const int*   idx   = (const int*)d_in[4];
  const unsigned char* rmask = (const unsigned char*)d_in[5];
  float* out = (float*)d_out;

  // ws layout (16-bit units first, then f32 scratch, then int nbr)
  unsigned short* u = (unsigned short*)d_ws;
  unsigned short* frag    = u;                       // 140288 bf16
  unsigned short* s0_bf   = u + 140288;              // 196608 bf16
  unsigned short* node_bf = u + 336896;              // 24576 bf16
  float* wsf = (float*)(u + 361472);                 // byte offset 722944
  float* node      = wsf;                            // 24576
  float* Dsel      = wsf + 24576;                    // 98304
  float* partial   = wsf + 122880;                   // 294912
  float* state_out = wsf + 417792;                   // 12288
  int*   nbr       = (int*)(wsf + 430080);           // 98304 ints

  struct PD { int widx; int off; int K, N, Kpad, Npad; };
  const PD pds[15] = {
      {6,       0, 272,  32, 288,  32},  // W_en
      {8,    9216,  32,  64,  32,  64},  // Wn1
      {10,  11264,  64,  32,  64,  32},  // Wn2
      {12,  13312, 193,  32, 224,  32},  // W_ee
      {14,  20480,  32,  64,  32,  64},  // We1
      {16,  22528,  64,  32,  64,  32},  // We2
      {18,  24576,  96,  64,  96,  64},  // Wm1
      {20,  30720,  64,  40,  64,  48},  // Wm2
      {24,  33792, 256, 128, 256, 128},  // Ws0
      {26,  66560,  16, 128,  32, 128},  // Wsi
      {28,  70656, 128, 128, 128, 128},  // Wp1
      {30,  87040, 128, 128, 128, 128},  // Wp2
      {32, 103424, 128, 128, 128, 128},  // Wp3
      {34, 119808, 128, 128, 128, 128},  // Wp4
      {36, 136192, 128,  20, 128,  32},  // Wout
  };
  for (int p = 0; p < 15; ++p) {
    int total = pds[p].Kpad * pds[p].Npad;
    pack_b_kernel<<<(total + 255) / 256, 256, 0, stream>>>(
        (const float*)d_in[pds[p].widx], frag + pds[p].off,
        pds[p].K, pds[p].N, pds[p].Kpad, pds[p].Npad);
  }

  node_kernel<<<48, 32, 0, stream>>>(
      msa, state, frag + 0, frag + 9216, frag + 11264,
      (const float*)d_in[7], (const float*)d_in[9], (const float*)d_in[11],
      s0_bf, node, node_bf);

  topk_kernel<<<LSEQ, 256, 0, stream>>>(xyz, nbr, Dsel);

  edge_msg_kernel<<<dim3(LSEQ, 8), 32, 0, stream>>>(
      pair, idx, xyz, node_bf, nbr, Dsel,
      frag + 13312, frag + 20480, frag + 22528, frag + 24576, frag + 30720,
      (const float*)d_in[13], (const float*)d_in[15], (const float*)d_in[17],
      (const float*)d_in[19], (const float*)d_in[21], partial);

  finalize_kernel<<<3, 256, 0, stream>>>(
      partial, node, xyz, rmask,
      (const float*)d_in[22], (const float*)d_in[23], state_out, out);

  si_kernel<<<48, 32, 0, stream>>>(
      s0_bf, state_out,
      frag + 33792, frag + 66560, frag + 70656, frag + 87040,
      frag + 103424, frag + 119808, frag + 136192,
      (const float*)d_in[25], (const float*)d_in[27],
      (const float*)d_in[29], (const float*)d_in[31], (const float*)d_in[33],
      (const float*)d_in[35], (const float*)d_in[37], out);
}
